// DeformableTransformerDecoder_63556926046234
// MI455X (gfx1250) — compile-verified, matrix operand order checked by reading, not run on hardware
//
#include <hip/hip_runtime.h>

// ---------------------------------------------------------------------------
// Deformable transformer decoder, CDNA5 (gfx1250) wave32 + bf16 WMMA.
// Global->LDS staging uses 128-bit loads (float4); LDS fragment reads are
// ds_load_b128; matrix math is v_wmma_f32_16x16x32_bf16 with fp32 accum.
// ---------------------------------------------------------------------------

#define B_    4
#define Q_    1024
#define DIM_  256
#define HEADS_ 8
#define DH_   64
#define INNER_ 512
#define DFF_  1024
#define TOT_  21504

typedef __attribute__((ext_vector_type(4)))  float  v4f;
typedef __attribute__((ext_vector_type(8)))  float  v8f;
typedef __attribute__((ext_vector_type(4)))  unsigned short us4;
typedef __attribute__((ext_vector_type(8)))  __bf16 v8bf;
typedef __attribute__((ext_vector_type(16))) __bf16 v16bf;

__device__ __forceinline__ unsigned short f2bf(float f) {
  unsigned int u = __float_as_uint(f);
  u += 0x7FFFu + ((u >> 16) & 1u);   // round-to-nearest-even
  return (unsigned short)(u >> 16);
}

__device__ __forceinline__ us4 f2bf4(v4f v) {
  us4 r;
  r[0] = f2bf(v[0]); r[1] = f2bf(v[1]);
  r[2] = f2bf(v[2]); r[3] = f2bf(v[3]);
  return r;
}

// Load one 16x32 bf16 operand fragment (A-layout per ISA 7.12.2) from an LDS
// row whose K-values are contiguous. Lane (l&15) = row, (l>>4) picks the K
// half: elements 0..7 = K base..base+7, elements 8..15 = K base+16..base+23,
// base = (l>>4)*8.  Two 16-byte ds_load_b128 per fragment.
__device__ __forceinline__ v16bf ld_frag(const unsigned short* rowp, int hl) {
  const v8bf* p0 = reinterpret_cast<const v8bf*>(rowp + hl * 8);
  const v8bf* p1 = reinterpret_cast<const v8bf*>(rowp + hl * 8 + 16);
  v8bf lo = *p0;
  v8bf hi = *p1;
  return __builtin_shufflevector(lo, hi, 0, 1, 2, 3, 4, 5, 6, 7,
                                 8, 9, 10, 11, 12, 13, 14, 15);
}

__device__ __forceinline__ float gelu_exact(float x) {
  return 0.5f * x * (1.0f + erff(x * 0.70710678118654752440f));
}

// ---------------------------------------------------------------------------
// Generic GEMM:  C[M,N] = act(A[M,K] @ W[K,N] + bias) (+ resid)
// block = 256 threads = 8 waves; C tile 128x64; each wave owns 16 rows x 64.
// Requires: K % 32 == 0, N % 4 == 0 (true for every call site).
// ---------------------------------------------------------------------------
template <int ACT>
__global__ __launch_bounds__(256) void gemm_bf16_wmma(
    const float* __restrict__ A, const float* __restrict__ W,
    const float* __restrict__ bias, const float* __restrict__ resid,
    float* __restrict__ C, int M, int N, int K) {
  __shared__ alignas(16) unsigned short sA[128][40];  // [m][k], 80B row stride
  __shared__ alignas(16) unsigned short sB[64][40];   // [n][k] (transposed)

  const int tid  = threadIdx.x;
  const int wave = tid >> 5;
  const int lane = tid & 31;
  const int hl   = lane >> 4;
  const int lr   = lane & 15;
  const int m0   = blockIdx.y * 128;
  const int n0   = blockIdx.x * 64;

  v8f acc[4] = {};

  for (int k0 = 0; k0 < K; k0 += 32) {
    // A tile 128x32: 1024 float4 loads (global_load_b128), b64 LDS stores
#pragma unroll
    for (int r = 0; r < 4; ++r) {
      int idx = tid + r * 256;
      int m = idx >> 3, k4 = idx & 7;
      int gm = m0 + m;
      v4f v = {};
      if (gm < M)
        v = *reinterpret_cast<const v4f*>(&A[(size_t)gm * K + k0 + k4 * 4]);
      *reinterpret_cast<us4*>(&sA[m][k4 * 4]) = f2bf4(v);
    }
    // B tile 32x64 -> [n][k]: 512 float4 loads, transposed scalar stores
#pragma unroll
    for (int r = 0; r < 2; ++r) {
      int idx = tid + r * 256;
      int kk = idx >> 4, n4 = idx & 15;
      int gn = n0 + n4 * 4, gk = k0 + kk;
      v4f v = {};
      if (gn < N)
        v = *reinterpret_cast<const v4f*>(&W[(size_t)gk * N + gn]);
      sB[n4 * 4 + 0][kk] = f2bf(v[0]);
      sB[n4 * 4 + 1][kk] = f2bf(v[1]);
      sB[n4 * 4 + 2][kk] = f2bf(v[2]);
      sB[n4 * 4 + 3][kk] = f2bf(v[3]);
    }
    __syncthreads();

    if (k0 + 32 < K) {                             // hint next K-slice of A
      int gm = m0 + (tid >> 1);
      if (gm < M) __builtin_prefetch(&A[(size_t)gm * K + k0 + 32], 0, 1);
    }

    v16bf aF = ld_frag(&sA[wave * 16 + lr][0], hl);
#pragma unroll
    for (int j = 0; j < 4; ++j) {
      v16bf bF = ld_frag(&sB[j * 16 + lr][0], hl);
      acc[j] = __builtin_amdgcn_wmma_f32_16x16x32_bf16(
          false, aF, false, bF, (short)0, acc[j], false, false);
    }
    __syncthreads();
  }

#pragma unroll
  for (int j = 0; j < 4; ++j) {
    int col = n0 + j * 16 + lr;
    if (col < N) {
      float bv = bias ? bias[col] : 0.0f;
#pragma unroll
      for (int v = 0; v < 8; ++v) {
        int row = m0 + wave * 16 + v + 8 * hl;
        if (row < M) {
          float x = acc[j][v] + bv;
          if (ACT == 1) x = gelu_exact(x);
          if (resid) x += resid[(size_t)row * N + col];
          C[(size_t)row * N + col] = x;
        }
      }
    }
  }
}

// ---------------------------------------------------------------------------
// Flash self-attention. qkv: (B,Q,1536) = [Q|K|V] each (HEADS*DH).
// grid = (B*HEADS, Q/128); block = 256 (8 waves). Each wave: 16 query rows.
// out: (B,Q,INNER), col = h*DH + d.
// ---------------------------------------------------------------------------
__global__ __launch_bounds__(256) void attn_flash(
    const float* __restrict__ qkv, float* __restrict__ out) {
  __shared__ alignas(16) unsigned short sQ[128][72];   // [q][d]
  __shared__ alignas(16) unsigned short sK[64][72];    // [key][d]
  __shared__ alignas(16) unsigned short sVt[64][72];   // [d][key]
  __shared__ alignas(16) unsigned short sP[8][16][72]; // per-wave P staging

  const int tid  = threadIdx.x;
  const int wave = tid >> 5;
  const int lane = tid & 31;
  const int hl   = lane >> 4;
  const int lr   = lane & 15;
  const int bh   = blockIdx.x;   // b*HEADS + h
  const int b    = bh >> 3;
  const int h    = bh & 7;
  const int qblk = blockIdx.y;   // 128-query block

  const size_t baseB = (size_t)b * Q_ * 1536;

  // Q block (pre-scaled by DH^-0.5 = 0.125): 2048 float4 loads
#pragma unroll
  for (int r = 0; r < 8; ++r) {
    int idx = tid + r * 256;
    int qr = idx >> 4, d4 = idx & 15;
    int gq = qblk * 128 + qr;
    v4f v = *reinterpret_cast<const v4f*>(
        &qkv[baseB + (size_t)gq * 1536 + h * 64 + d4 * 4]);
    v *= 0.125f;
    *reinterpret_cast<us4*>(&sQ[qr][d4 * 4]) = f2bf4(v);
  }

  float mi[8], li[8];
#pragma unroll
  for (int v = 0; v < 8; ++v) { mi[v] = -1e30f; li[v] = 0.0f; }
  v8f oA[4] = {};

  for (int kt = 0; kt < 16; ++kt) {
    __syncthreads();
#pragma unroll
    for (int r = 0; r < 4; ++r) {     // K tile + V tile (transposed)
      int idx = tid + r * 256;
      int kr = idx >> 4, d4 = idx & 15;
      int gk = kt * 64 + kr;
      size_t base = baseB + (size_t)gk * 1536 + h * 64 + d4 * 4;
      v4f kv = *reinterpret_cast<const v4f*>(&qkv[base + 512]);
      v4f vv = *reinterpret_cast<const v4f*>(&qkv[base + 1024]);
      *reinterpret_cast<us4*>(&sK[kr][d4 * 4]) = f2bf4(kv);
      sVt[d4 * 4 + 0][kr] = f2bf(vv[0]);
      sVt[d4 * 4 + 1][kr] = f2bf(vv[1]);
      sVt[d4 * 4 + 2][kr] = f2bf(vv[2]);
      sVt[d4 * 4 + 3][kr] = f2bf(vv[3]);
    }
    __syncthreads();

    // S = Q @ K^T  (16 rows x 64 keys per wave)
    v8f s[4] = {};
#pragma unroll
    for (int ks = 0; ks < 2; ++ks) {
      v16bf aF = ld_frag(&sQ[wave * 16 + lr][ks * 32], hl);
#pragma unroll
      for (int j = 0; j < 4; ++j) {
        v16bf bF = ld_frag(&sK[j * 16 + lr][ks * 32], hl);
        s[j] = __builtin_amdgcn_wmma_f32_16x16x32_bf16(
            false, aF, false, bF, (short)0, s[j], false, false);
      }
    }

    // online softmax: row = v + 8*hl, col = j*16 + lr; reduce across columns
    float pm[8], nm[8], corr[8], ps[8];
#pragma unroll
    for (int v = 0; v < 8; ++v) {
      pm[v] = fmaxf(fmaxf(s[0][v], s[1][v]), fmaxf(s[2][v], s[3][v]));
      pm[v] = fmaxf(pm[v], __shfl_xor(pm[v], 1, 32));
      pm[v] = fmaxf(pm[v], __shfl_xor(pm[v], 2, 32));
      pm[v] = fmaxf(pm[v], __shfl_xor(pm[v], 4, 32));
      pm[v] = fmaxf(pm[v], __shfl_xor(pm[v], 8, 32));
      nm[v]   = fmaxf(mi[v], pm[v]);
      corr[v] = __expf(mi[v] - nm[v]);
      mi[v]   = nm[v];
      ps[v]   = 0.0f;
    }
#pragma unroll
    for (int j = 0; j < 4; ++j) {
#pragma unroll
      for (int v = 0; v < 8; ++v) {
        float p = __expf(s[j][v] - nm[v]);
        s[j][v] = p;
        ps[v] += p;
      }
    }
#pragma unroll
    for (int v = 0; v < 8; ++v) {
      ps[v] += __shfl_xor(ps[v], 1, 32);
      ps[v] += __shfl_xor(ps[v], 2, 32);
      ps[v] += __shfl_xor(ps[v], 4, 32);
      ps[v] += __shfl_xor(ps[v], 8, 32);
      li[v] = li[v] * corr[v] + ps[v];
    }
    // rescale O and stage P (C-layout -> A-layout via wave-private LDS)
#pragma unroll
    for (int j = 0; j < 4; ++j) {
#pragma unroll
      for (int v = 0; v < 8; ++v) {
        oA[j][v] *= corr[v];
        sP[wave][v + 8 * hl][j * 16 + lr] = f2bf(s[j][v]);
      }
    }
    // O += P @ V
#pragma unroll
    for (int ks = 0; ks < 2; ++ks) {
      v16bf aP = ld_frag(&sP[wave][lr][ks * 32], hl);
#pragma unroll
      for (int j = 0; j < 4; ++j) {
        v16bf bF = ld_frag(&sVt[j * 16 + lr][ks * 32], hl);
        oA[j] = __builtin_amdgcn_wmma_f32_16x16x32_bf16(
            false, aP, false, bF, (short)0, oA[j], false, false);
      }
    }
  }

#pragma unroll
  for (int j = 0; j < 4; ++j) {
    int col = h * 64 + j * 16 + lr;
#pragma unroll
    for (int v = 0; v < 8; ++v) {
      int q = qblk * 128 + wave * 16 + v + 8 * hl;
      out[((size_t)b * Q_ + q) * INNER_ + col] = oA[j][v] / li[v];
    }
  }
}

// ---------------------------------------------------------------------------
// LayerNorm over last dim (=256): out = LN(in [+ addpre]) * w + b [+ addpost]
// one row per block of 256 threads.
// ---------------------------------------------------------------------------
__global__ __launch_bounds__(256) void layernorm256(
    const float* __restrict__ in, const float* __restrict__ addpre,
    const float* __restrict__ w, const float* __restrict__ bvec,
    const float* __restrict__ addpost, float* __restrict__ out) {
  __shared__ float red[256];
  __shared__ float stat[2];
  const int r = blockIdx.x, t = threadIdx.x;
  const size_t off = (size_t)r * 256 + t;
  float v = in[off];
  if (addpre) v += addpre[off];
  red[t] = v;
  __syncthreads();
  for (int s = 128; s > 0; s >>= 1) {
    if (t < s) red[t] += red[t + s];
    __syncthreads();
  }
  if (t == 0) stat[0] = red[0] * (1.0f / 256.0f);
  __syncthreads();
  float d = v - stat[0];
  red[t] = d * d;
  __syncthreads();
  for (int s = 128; s > 0; s >>= 1) {
    if (t < s) red[t] += red[t + s];
    __syncthreads();
  }
  if (t == 0) stat[1] = rsqrtf(red[0] * (1.0f / 256.0f) + 1e-5f);
  __syncthreads();
  float o = d * stat[1] * w[t] + bvec[t];
  if (addpost) o += addpost[off];
  out[off] = o;
}

// pos-MLP hidden: relu(center_pos @ pe1_w + pe1_b), one row / block
__global__ __launch_bounds__(256) void pos_hidden_kernel(
    const float* __restrict__ cp, const float* __restrict__ pe1w,
    const float* __restrict__ pe1b, float* __restrict__ hid) {
  const int r = blockIdx.x, d = threadIdx.x;
  float cx = cp[(size_t)r * 2 + 0];
  float cy = cp[(size_t)r * 2 + 1];
  float v = cx * pe1w[d] + cy * pe1w[256 + d] + pe1b[d];
  hid[(size_t)r * 256 + d] = fmaxf(v, 0.0f);
}

// ---------------------------------------------------------------------------
// Deformable sampling: one wave per (b,q,h); lane covers d and d+32.
// value:(B,TOT,INNER)  off:(B,Q,432)  aw:(B,Q,216)  out:(B,Q,INNER)
// ---------------------------------------------------------------------------
__global__ __launch_bounds__(256) void deform_kernel(
    const float* __restrict__ value, const float* __restrict__ offb,
    const float* __restrict__ awb, const float* __restrict__ cp,
    float* __restrict__ out) {
  const int wave = threadIdx.x >> 5;
  const int lane = threadIdx.x & 31;
  const int idx = blockIdx.x * 8 + wave;      // 0 .. B*Q*HEADS-1
  const int b   = idx >> 13;
  const int rem = idx & 8191;
  const int q   = rem >> 3;
  const int h   = rem & 7;
  const int bq  = b * Q_ + q;

  const float* aw27 = awb + (size_t)bq * 216 + h * 27;
  float mx = -1e30f;
  for (int j = 0; j < 27; ++j) mx = fmaxf(mx, aw27[j]);
  float sum = 0.0f;
  for (int j = 0; j < 27; ++j) sum += __expf(aw27[j] - mx);
  const float inv = 1.0f / sum;

  const float cx = cp[(size_t)bq * 2 + 0];
  const float cy = cp[(size_t)bq * 2 + 1];
  const float* op = offb + (size_t)bq * 432 + h * 54;

  float a0 = 0.0f, a1 = 0.0f;
  const int LH[3]  = {128, 64, 32};
  const int LST[3] = {0, 16384, 20480};
  for (int l = 0; l < 3; ++l) {
    const int Hi = LH[l], Wi = LH[l], st = LST[l];
    const float Hf = (float)Hi, Wf = (float)Wi;
    for (int pt = 0; pt < 9; ++pt) {
      const int lp = l * 9 + pt;
      float ox = op[lp * 2 + 0], oy = op[lp * 2 + 1];
      float px = (cx + ox / Wf) * Wf - 0.5f;
      float py = (cy + oy / Hf) * Hf - 0.5f;
      float x0 = floorf(px), y0 = floorf(py);
      float fx = px - x0, fy = py - y0;
      int xi = (int)x0, yi = (int)y0;
      float wt = __expf(aw27[lp] - mx) * inv;
      for (int c = 0; c < 4; ++c) {
        int dx = c & 1, dy = c >> 1;
        int xx = xi + dx, yy = yi + dy;
        if (xx >= 0 && xx < Wi && yy >= 0 && yy < Hi) {
          float cw = (dx ? fx : 1.0f - fx) * (dy ? fy : 1.0f - fy) * wt;
          const float* vp =
              value + ((size_t)b * TOT_ + st + yy * Wi + xx) * INNER_ + h * 64;
          a0 += cw * vp[lane];
          a1 += cw * vp[lane + 32];
        }
      }
    }
  }
  float* o = out + (size_t)bq * INNER_ + h * 64;
  o[lane]      = a0;
  o[lane + 32] = a1;
}

// ---------------------------------------------------------------------------
extern "C" void kernel_launch(void* const* d_in, const int* in_sizes, int n_in,
                              void* d_out, int out_size, void* d_ws,
                              size_t ws_size, hipStream_t stream) {
  const float* x_in     = (const float*)d_in[0];
  const float* src      = (const float*)d_in[1];
  const float* cp       = (const float*)d_in[2];
  const float* ln_sa_w  = (const float*)d_in[5];
  const float* ln_sa_b  = (const float*)d_in[6];
  const float* qkv_w    = (const float*)d_in[7];
  const float* sa_out_w = (const float*)d_in[8];
  const float* sa_out_b = (const float*)d_in[9];
  const float* ln_ca_w  = (const float*)d_in[10];
  const float* ln_ca_b  = (const float*)d_in[11];
  const float* off_w    = (const float*)d_in[12];
  const float* off_b    = (const float*)d_in[13];
  const float* aw_w     = (const float*)d_in[14];
  const float* aw_b     = (const float*)d_in[15];
  const float* val_w    = (const float*)d_in[16];
  const float* val_b    = (const float*)d_in[17];
  const float* out_w    = (const float*)d_in[18];
  const float* out_b    = (const float*)d_in[19];
  const float* ln_ff_w  = (const float*)d_in[20];
  const float* ln_ff_b  = (const float*)d_in[21];
  const float* ff1_w    = (const float*)d_in[22];
  const float* ff1_b    = (const float*)d_in[23];
  const float* ff2_w    = (const float*)d_in[24];
  const float* ff2_b    = (const float*)d_in[25];
  const float* pe1_w    = (const float*)d_in[26];
  const float* pe1_b    = (const float*)d_in[27];
  const float* pe2_w    = (const float*)d_in[28];
  const float* pe2_b    = (const float*)d_in[29];

  const int BQ = B_ * Q_;     // 4096
  const int BT = B_ * TOT_;   // 86016

  float* p = (float*)d_ws;
  float* xbuf = p; p += (size_t)BQ * DIM_;
  float* pos  = p; p += (size_t)BQ * DIM_;
  float* hbuf = p; p += (size_t)BQ * DIM_;
  float* xq   = p; p += (size_t)BQ * DIM_;
  float* qkvb = p; p += (size_t)BQ * 1536;
  float* attn = p; p += (size_t)BQ * INNER_;
  float* offb = p; p += (size_t)BQ * 432;
  float* awb  = p; p += (size_t)BQ * 216;
  float* samp = p; p += (size_t)BQ * INNER_;
  float* ffh  = p; p += (size_t)BQ * DFF_;
  float* srcn = p; p += (size_t)BT * DIM_;
  float* valb = p; p += (size_t)BT * INNER_;

  auto gemm = [&](const float* A, const float* W, const float* bias,
                  const float* resid, float* C, int M, int N, int K, int act) {
    dim3 g((N + 63) / 64, (M + 127) / 128);
    if (act)
      gemm_bf16_wmma<1><<<g, 256, 0, stream>>>(A, W, bias, resid, C, M, N, K);
    else
      gemm_bf16_wmma<0><<<g, 256, 0, stream>>>(A, W, bias, resid, C, M, N, K);
  };

  hipMemcpyAsync(xbuf, x_in, (size_t)BQ * DIM_ * sizeof(float),
                 hipMemcpyDeviceToDevice, stream);

  // pos = relu(cp @ pe1 + b1) @ pe2 + b2
  pos_hidden_kernel<<<dim3(BQ), 256, 0, stream>>>(cp, pe1_w, pe1_b, hbuf);
  gemm(hbuf, pe2_w, pe2_b, nullptr, pos, BQ, DIM_, DIM_, 0);

  for (int i = 0; i < 2; ++i) {
    // ---- self-attention ----
    layernorm256<<<dim3(BQ), 256, 0, stream>>>(
        xbuf, pos, ln_sa_w + i * DIM_, ln_sa_b + i * DIM_, nullptr, hbuf);
    gemm(hbuf, qkv_w + (size_t)i * DIM_ * 1536, nullptr, nullptr, qkvb,
         BQ, 1536, DIM_, 0);
    attn_flash<<<dim3(B_ * HEADS_, Q_ / 128), 256, 0, stream>>>(qkvb, attn);
    gemm(attn, sa_out_w + (size_t)i * INNER_ * DIM_, sa_out_b + i * DIM_,
         xbuf, xbuf, BQ, DIM_, INNER_, 0);

    // ---- deformable cross-attention ----
    layernorm256<<<dim3(BQ), 256, 0, stream>>>(
        xbuf, nullptr, ln_ca_w + i * DIM_, ln_ca_b + i * DIM_, pos, xq);
    layernorm256<<<dim3(BT), 256, 0, stream>>>(
        src, nullptr, ln_ca_w + i * DIM_, ln_ca_b + i * DIM_, nullptr, srcn);
    gemm(srcn, val_w + (size_t)i * DIM_ * INNER_, val_b + i * INNER_, nullptr,
         valb, BT, INNER_, DIM_, 0);
    gemm(xq, off_w + (size_t)i * DIM_ * 432, off_b + i * 432, nullptr, offb,
         BQ, 432, DIM_, 0);
    gemm(xq, aw_w + (size_t)i * DIM_ * 216, aw_b + i * 216, nullptr, awb,
         BQ, 216, DIM_, 0);
    deform_kernel<<<dim3(BQ * HEADS_ / 8), 256, 0, stream>>>(
        valb, offb, awb, cp, samp);
    gemm(samp, out_w + (size_t)i * INNER_ * DIM_, out_b + i * DIM_, xbuf,
         xbuf, BQ, DIM_, INNER_, 0);

    // ---- feed-forward ----
    layernorm256<<<dim3(BQ), 256, 0, stream>>>(
        xbuf, nullptr, ln_ff_w + i * DIM_, ln_ff_b + i * DIM_, nullptr, hbuf);
    gemm(hbuf, ff1_w + (size_t)i * DIM_ * DFF_, ff1_b + i * DFF_, nullptr,
         ffh, BQ, DFF_, DIM_, 1);  // GELU fused
    gemm(ffh, ff2_w + (size_t)i * DFF_ * DIM_, ff2_b + i * DIM_, xbuf, xbuf,
         BQ, DIM_, DFF_, 0);
  }

  hipMemcpyAsync(d_out, xbuf, (size_t)BQ * DIM_ * sizeof(float),
                 hipMemcpyDeviceToDevice, stream);
}
